// Fractal_GNN_no_rel_20796231647838
// MI455X (gfx1250) — matrix operand order repaired and checked by reading, not compiled, not run on hardware
//
#include <hip/hip_runtime.h>
#include <hip/hip_bf16.h>
#include <stdint.h>

typedef __attribute__((ext_vector_type(16))) _Float16 v16h;
typedef __attribute__((ext_vector_type(8)))  float    v8f;

#define WMMA_F16(a,b,c) __builtin_amdgcn_wmma_f32_16x16x32_f16(false,(a),false,(b),(short)0,(c),false,false)

// ---------------- fragment helpers ----------------
// A-matrix 16x32 f16 layout (ISA 7.12.2): lane m=lane&15, hs=lane>>4.
// VGPR v=0..3 -> K = hs*8 + 2v,2v+1 ; v=4..7 -> K = 16 + hs*8 + 2(v-4),+1.
// => per lane: two contiguous 8-float chunks at row[kb*32 + hs*8] and row[kb*32 + 16 + hs*8].
__device__ __forceinline__ v16h load_a_frag(const float* __restrict__ row, int kb, int hs) {
  const float4* p0 = (const float4*)(row + kb * 32 + hs * 8);
  const float4* p1 = (const float4*)(row + kb * 32 + 16 + hs * 8);
  float4 a0 = p0[0], a1 = p0[1], b0 = p1[0], b1 = p1[1];
  v16h r;
  r[0] = (_Float16)a0.x; r[1] = (_Float16)a0.y; r[2]  = (_Float16)a0.z; r[3]  = (_Float16)a0.w;
  r[4] = (_Float16)a1.x; r[5] = (_Float16)a1.y; r[6]  = (_Float16)a1.z; r[7]  = (_Float16)a1.w;
  r[8] = (_Float16)b0.x; r[9] = (_Float16)b0.y; r[10] = (_Float16)b0.z; r[11] = (_Float16)b0.w;
  r[12]= (_Float16)b1.x; r[13]= (_Float16)b1.y; r[14] = (_Float16)b1.z; r[15] = (_Float16)b1.w;
  return r;
}

// Pre-swizzled B fragment: frag (kb,nb) = 256 dwords, lane-major (8 dwords = 32B per lane).
__device__ __forceinline__ v16h load_b_frag(const uint32_t* __restrict__ wmat, int kb, int nb, int lane) {
  return ((const v16h*)(wmat + (size_t)(kb * 4 + nb) * 256))[lane];
}

// ---------------- weight conversion (f32 row-major [K][64] -> f16 B fragments) ----------------
// B-matrix 32x16 f16 layout (ISA 7.12.4/5 striping): n = nb*16 + (lane&15), hs=lane>>4,
// VGPR v=0..3 -> K = kb*32 + hs*8 + 2v, 2v+1 ; v=4..7 -> K = kb*32 + 16 + hs*8 + 2(v-4), +1.
__global__ void cvt_weights_kernel(const float* __restrict__ src, uint32_t* __restrict__ dst,
                                   int nmat, int K) {
  int fpm = (K / 32) * 4;           // fragments per matrix
  int dpm = fpm * 256;              // dwords per matrix
  int tid = blockIdx.x * blockDim.x + threadIdx.x;
  if (tid >= nmat * dpm) return;
  int m = tid / dpm, r = tid % dpm;
  int fid = r >> 8, lane = (r >> 3) & 31, v = r & 7;
  int kb = fid >> 2, nb = fid & 3;
  int n = nb * 16 + (lane & 15), hs = lane >> 4;
  int k0 = kb * 32 + hs * 8 + 2 * v + (v >= 4 ? 8 : 0);
  const float* W = src + (size_t)m * K * 64;
  _Float16* out = (_Float16*)dst;
  out[(size_t)tid * 2 + 0] = (_Float16)W[(size_t)k0 * 64 + n];
  out[(size_t)tid * 2 + 1] = (_Float16)W[(size_t)(k0 + 1) * 64 + n];
}

// ---------------- embed: h = x @ embed_W + embed_b ----------------
__global__ void embed_kernel(float* __restrict__ h, const float* __restrict__ x,
                             const uint32_t* __restrict__ we, const float* __restrict__ eb,
                             int ntiles) {
  int wave = blockIdx.x * (blockDim.x >> 5) + (threadIdx.x >> 5);
  int lane = threadIdx.x & 31;
  if (wave >= ntiles) return;
  int m = lane & 15, hs = lane >> 4;
  int node = wave * 16 + m;
  v16h a = load_a_frag(x + (size_t)node * 32, 0, hs);   // FIN=32 -> one K-block
#pragma unroll
  for (int nb = 0; nb < 4; ++nb) {
    float bv = eb[nb * 16 + m];
    v8f c;
#pragma unroll
    for (int j = 0; j < 8; ++j) c[j] = bv;
    c = WMMA_F16(a, load_b_frag(we, 0, nb, lane), c);
#pragma unroll
    for (int j = 0; j < 8; ++j) {
      int row = wave * 16 + j + hs * 8;                 // C layout: VGPR j -> M=j (+8 for hi lanes)
      h[(size_t)row * 64 + nb * 16 + m] = c[j];
    }
  }
}

// ---------------- per-conv scratch reset ----------------
__global__ void reset_conv_kernel(float* agg, int* deg, int* order, int* meta, int N) {
  int i = blockIdx.x * blockDim.x + threadIdx.x;
  if (i < N * 64) agg[i] = 0.f;
  if (i < N) deg[i] = 0;
  if (i < N + 256) order[i] = -1;
  if (i < 64) meta[i] = 0;
}

__global__ void deg_kernel(int* __restrict__ deg, const int* __restrict__ ei, int E) {
  int i = blockIdx.x * blockDim.x + threadIdx.x;
  if (i < E) atomicAdd(&deg[ei[E + i]], 1);
}

// agg[dst] += h[src]: one thread per (edge, 4-float chunk)
__global__ void agg_kernel(float* __restrict__ agg, const float* __restrict__ h,
                           const int* __restrict__ ei, int E) {
  int t = blockIdx.x * blockDim.x + threadIdx.x;
  if (t >= E * 16) return;
  int e = t >> 4, c = t & 15;
  int s = ei[e], d = ei[E + e];
  float4 v = ((const float4*)(h + (size_t)s * 64))[c];
  float* p = agg + (size_t)d * 64 + c * 4;
  unsafeAtomicAdd(p + 0, v.x);
  unsafeAtomicAdd(p + 1, v.y);
  unsafeAtomicAdd(p + 2, v.z);
  unsafeAtomicAdd(p + 3, v.w);
}

// meta layout: [0]=ntiles, [1..11]=counts, [12..22]=offsets, [23..33]=fill
__global__ void bucket_count_kernel(int* __restrict__ meta, const int* __restrict__ deg, int N) {
  int i = blockIdx.x * blockDim.x + threadIdx.x;
  if (i >= N) return;
  int dg = deg[i];
  if (dg > 0) atomicAdd(&meta[1 + min(dg, 10)], 1);
}

__global__ void scan_kernel(int* __restrict__ meta, int* __restrict__ tb) {
  if (blockIdx.x | threadIdx.x) return;
  int off = 0;
  for (int d = 0; d < 11; ++d) {
    meta[12 + d] = off;
    int pc = (meta[1 + d] + 15) & ~15;                  // pad bucket to 16 so tiles never mix buckets
    int t0 = off >> 4;
    for (int t = 0; t < (pc >> 4); ++t) tb[t0 + t] = d;
    off += pc;
  }
  meta[0] = off >> 4;
}

__global__ void bucket_fill_kernel(int* __restrict__ order, int* __restrict__ meta,
                                   const int* __restrict__ deg, int N) {
  int i = blockIdx.x * blockDim.x + threadIdx.x;
  if (i >= N) return;
  int dg = deg[i];
  if (dg > 0) {
    int d = min(dg, 10);
    int pos = atomicAdd(&meta[23 + d], 1);
    order[meta[12 + d] + pos] = i;
  }
}

// ---------------- grouped MFConv GEMM: out = agg@Wl[d] + bl[d] + h@Wr[d], per 16-node tile ----------------
__global__ void conv_gemm_kernel(float* __restrict__ h, const float* __restrict__ agg,
                                 const int* __restrict__ order, const int* __restrict__ tb,
                                 const int* __restrict__ meta,
                                 const uint32_t* __restrict__ wl, const uint32_t* __restrict__ wr,
                                 const float* __restrict__ bl) {
  int wave = blockIdx.x * (blockDim.x >> 5) + (threadIdx.x >> 5);
  int lane = threadIdx.x & 31;
  if (wave >= meta[0]) return;
  int d = tb[wave];
  int base = wave << 4;
  int m = lane & 15, hs = lane >> 4;
  int node = order[base + m];
  v16h aA0 = {}, aA1 = {}, aH0 = {}, aH1 = {};
  if (node >= 0) {
    const float* ra = agg + (size_t)node * 64;
    const float* rh = h + (size_t)node * 64;
    aA0 = load_a_frag(ra, 0, hs); aA1 = load_a_frag(ra, 1, hs);
    aH0 = load_a_frag(rh, 0, hs); aH1 = load_a_frag(rh, 1, hs);
  }
  const uint32_t* wld = wl + (size_t)d * 2048;
  const uint32_t* wrd = wr + (size_t)d * 2048;
  const float* bias = bl + d * 64;
#pragma unroll
  for (int nb = 0; nb < 4; ++nb) {
    float bv = bias[nb * 16 + m];
    v8f c;
#pragma unroll
    for (int j = 0; j < 8; ++j) c[j] = bv;
    c = WMMA_F16(aA0, load_b_frag(wld, 0, nb, lane), c);
    c = WMMA_F16(aA1, load_b_frag(wld, 1, nb, lane), c);
    c = WMMA_F16(aH0, load_b_frag(wrd, 0, nb, lane), c);
    c = WMMA_F16(aH1, load_b_frag(wrd, 1, nb, lane), c);
#pragma unroll
    for (int j = 0; j < 8; ++j) {
      int nd = order[base + j + hs * 8];
      if (nd >= 0) h[(size_t)nd * 64 + nb * 16 + m] = c[j];   // in-place ok: A frags preloaded
    }
  }
}

__global__ void relu_kernel(float* h, int n) {
  int i = blockIdx.x * blockDim.x + threadIdx.x;
  if (i < n) h[i] = fmaxf(h[i], 0.f);
}

__global__ void zero_kernel(float* p, int n) {
  int i = blockIdx.x * blockDim.x + threadIdx.x;
  if (i < n) p[i] = 0.f;
}

// pooled[batch_idx[i]] += h[i] over ground nodes (i < NG by construction)
__global__ void pool_kernel(float* __restrict__ pooled, const float* __restrict__ h,
                            const int* __restrict__ bidx, int NG) {
  int t = blockIdx.x * blockDim.x + threadIdx.x;
  if (t >= NG * 64) return;
  int node = t >> 6, f = t & 63;
  unsafeAtomicAdd(&pooled[bidx[node] * 64 + f], h[(size_t)node * 64 + f]);
}

__global__ void mlp_kernel(float* __restrict__ out, const float* __restrict__ pooled,
                           const float* __restrict__ W1, const float* __restrict__ b1,
                           const float* __restrict__ W2, const float* __restrict__ b2) {
  int b = threadIdx.x;
  if (b >= 64) return;
  float acc = 0.f;
  for (int j = 0; j < 64; ++j) {
    float z = b1[j];
    for (int k = 0; k < 64; ++k) z += pooled[b * 64 + k] * W1[k * 64 + j];
    acc += fmaxf(z, 0.f) * W2[j];
  }
  out[b] = acc + b2[0];
}

// ---------------- host orchestration ----------------
extern "C" void kernel_launch(void* const* d_in, const int* in_sizes, int n_in,
                              void* d_out, int out_size, void* d_ws, size_t ws_size,
                              hipStream_t stream) {
  const int N = 65536, NG = 32768, B = 64;
  const int EG = in_sizes[1] / 2, ES = in_sizes[2] / 2, EX = in_sizes[3] / 2;

  const float* x        = (const float*)d_in[0];
  const int* ei_ground  = (const int*)d_in[1];
  const int* ei_sub     = (const int*)d_in[2];
  const int* ei_g2s     = (const int*)d_in[3];
  const int* ei_s2g     = (const int*)d_in[4];
  const int* batch_idx  = (const int*)d_in[7];
  const float* embed_W  = (const float*)d_in[8];
  const float* embed_b  = (const float*)d_in[9];
  const float* Wl_in[4] = {(const float*)d_in[10], (const float*)d_in[13], (const float*)d_in[16], (const float*)d_in[19]};
  const float* bl_in[4] = {(const float*)d_in[11], (const float*)d_in[14], (const float*)d_in[17], (const float*)d_in[20]};
  const float* Wr_in[4] = {(const float*)d_in[12], (const float*)d_in[15], (const float*)d_in[18], (const float*)d_in[21]};
  const float* W1 = (const float*)d_in[22];
  const float* b1 = (const float*)d_in[23];
  const float* W2 = (const float*)d_in[24];
  const float* b2 = (const float*)d_in[25];

  char* base = (char*)d_ws;
  size_t off = 0;
  auto carve = [&](size_t bytes) -> void* {
    void* p = base + off;
    off = (off + bytes + 255) & ~(size_t)255;
    return p;
  };
  float* h      = (float*)carve((size_t)N * 64 * 4);
  float* agg    = (float*)carve((size_t)N * 64 * 4);
  int* deg      = (int*)carve((size_t)N * 4);
  int* order    = (int*)carve((size_t)(N + 256) * 4);
  int* tb       = (int*)carve((size_t)(N / 16 + 64) * 4);
  int* meta     = (int*)carve(64 * 4);
  float* pooled = (float*)carve((size_t)B * 64 * 4);
  uint32_t* wcvt_embed = (uint32_t*)carve(1024 * 4);
  uint32_t* wcvt[8];
  for (int i = 0; i < 8; ++i) wcvt[i] = (uint32_t*)carve((size_t)22 * 2048 * 4);

  // 1) convert weights to WMMA-native f16 fragments (per-call: weights are inputs)
  cvt_weights_kernel<<<(1024 + 255) / 256, 256, 0, stream>>>(embed_W, wcvt_embed, 1, 32);
  for (int nm = 0; nm < 4; ++nm) {
    cvt_weights_kernel<<<(22 * 2048 + 255) / 256, 256, 0, stream>>>(Wl_in[nm], wcvt[nm * 2 + 0], 22, 64);
    cvt_weights_kernel<<<(22 * 2048 + 255) / 256, 256, 0, stream>>>(Wr_in[nm], wcvt[nm * 2 + 1], 22, 64);
  }

  // 2) embed
  {
    int tiles = N / 16;
    embed_kernel<<<(tiles + 7) / 8, 256, 0, stream>>>(h, x, wcvt_embed, embed_b, tiles);
  }

  // 3) conv layers: ground, g2s, sub, s2g per layer (wcvt index matches this order)
  const int* eis[4] = {ei_ground, ei_g2s, ei_sub, ei_s2g};
  const int  Es[4]  = {EG, EX, ES, EX};
  const int maxTiles = N / 16 + 11;
  for (int l = 0; l < 2; ++l) {
    if (l) relu_kernel<<<(N * 64 + 255) / 256, 256, 0, stream>>>(h, N * 64);
    for (int cv = 0; cv < 4; ++cv) {
      int E = Es[cv];
      const int* ei = eis[cv];
      reset_conv_kernel<<<(N * 64 + 255) / 256, 256, 0, stream>>>(agg, deg, order, meta, N);
      deg_kernel<<<(E + 255) / 256, 256, 0, stream>>>(deg, ei, E);
      agg_kernel<<<(E * 16 + 255) / 256, 256, 0, stream>>>(agg, h, ei, E);
      bucket_count_kernel<<<(N + 255) / 256, 256, 0, stream>>>(meta, deg, N);
      scan_kernel<<<1, 32, 0, stream>>>(meta, tb);
      bucket_fill_kernel<<<(N + 255) / 256, 256, 0, stream>>>(order, meta, deg, N);
      conv_gemm_kernel<<<(maxTiles + 7) / 8, 256, 0, stream>>>(
          h, agg, order, tb, meta,
          wcvt[cv * 2 + 0] + (size_t)l * 11 * 2048,
          wcvt[cv * 2 + 1] + (size_t)l * 11 * 2048,
          bl_in[cv] + (size_t)l * 11 * 64);
    }
  }

  // 4) pool over ground nodes + MLP head
  zero_kernel<<<(B * 64 + 255) / 256, 256, 0, stream>>>(pooled, B * 64);
  pool_kernel<<<(NG * 64 + 255) / 256, 256, 0, stream>>>(pooled, h, batch_idx, NG);
  mlp_kernel<<<1, 64, 0, stream>>>((float*)d_out, pooled, W1, b1, W2, b2);
}
